// SpatioTemporalBlock_7627861918136
// MI455X (gfx1250) — compile-verified, matrix operand order checked by reading, not run on hardware
//
#include <hip/hip_runtime.h>
#include <hip/hip_bf16.h>

// ---- problem constants (match reference) ----
constexpr int B  = 32;
constexpr int N  = 512;
constexpr int F  = 32;
constexpr int T  = 24;
constexpr int GC = 64;
constexpr int TC = 64;
constexpr int KK = 512;          // GEMM reduction depth (== N)

typedef __bf16 bf16;
typedef bf16  bf16x8  __attribute__((ext_vector_type(8)));
typedef bf16  bf16x16 __attribute__((ext_vector_type(16)));
typedef float f32x8   __attribute__((ext_vector_type(8)));

#define LDP 40   // LDS row pitch in bf16 (80 B; keeps 16B alignment, dodges bank conflicts)

// =====================================================================
// CDNA5 async global->LDS copy (ASYNCcnt-tracked DMA), ISA §15.18.3 op 98
// =====================================================================
__device__ inline void async_copy16(unsigned lds_off, const bf16* g) {
  asm volatile("global_load_async_to_lds_b128 %0, %1, off"
               :: "v"(lds_off), "v"((unsigned long long)(size_t)g)
               : "memory");
}
__device__ inline void wait_async0() {
  asm volatile("s_wait_asynccnt 0x0" ::: "memory");
}

// =====================================================================
// Small temporal-attention kernels (negligible FLOPs)
// =====================================================================

// xw1[b,t,f] = sum_n x[b,n,f,t] * w1[n]
__global__ void k_xw1(const float* __restrict__ x, const float* __restrict__ w1,
                      float* __restrict__ xw1) {
  int i = blockIdx.x * blockDim.x + threadIdx.x;
  if (i >= B * T * F) return;
  int b = i / (T * F), r = i % (T * F), t = r / F, f = r % F;
  const float* xp = x + (size_t)b * N * F * T + f * T + t;
  float s = 0.f;
  for (int n = 0; n < N; ++n) s += xp[(size_t)n * F * T] * w1[n];
  xw1[i] = s;
}

// rhs_t[b,n,t] = sum_f x[b,n,f,t] * w3[f]
__global__ void k_rhs_t(const float* __restrict__ x, const float* __restrict__ w3,
                        float* __restrict__ rhs) {
  int i = blockIdx.x * blockDim.x + threadIdx.x;
  if (i >= B * N * T) return;
  int bn = i / T, t = i % T;
  const float* xp = x + (size_t)bn * F * T + t;
  float s = 0.f;
  for (int f = 0; f < F; ++f) s += xp[f * T] * w3[f];
  rhs[i] = s;
}

// lhs_t[b,t,n] = sum_f xw1[b,t,f] * w2[f,n]
__global__ void k_lhs_t(const float* __restrict__ xw1, const float* __restrict__ w2,
                        float* __restrict__ lhs) {
  int i = blockIdx.x * blockDim.x + threadIdx.x;
  if (i >= B * T * N) return;
  int bt = i / N, n = i % N;
  const float* xp = xw1 + (size_t)bt * F;
  float s = 0.f;
  for (int f = 0; f < F; ++f) s += xp[f] * w2[f * N + n];
  lhs[i] = s;
}

// prod[b,s,u] = sum_n lhs[b,s,n] * rhs[b,n,u]
__global__ void k_prod(const float* __restrict__ lhs, const float* __restrict__ rhs,
                       float* __restrict__ prod) {
  int i = blockIdx.x * blockDim.x + threadIdx.x;
  if (i >= B * T * T) return;
  int b = i / (T * T), r = i % (T * T), s = r / T, u = r % T;
  const float* lp = lhs + ((size_t)b * T + s) * N;
  const float* rp = rhs + (size_t)b * N * T + u;
  float acc = 0.f;
  for (int n = 0; n < N; ++n) acc += lp[n] * rp[(size_t)n * T];
  prod[i] = acc;
}

// E[b,s,u] = sum_t proj[s,t]*sigmoid(prod[b,t,u]+bias[t,u]); temp_w = softmax_s(E)
__global__ void k_tempw(const float* __restrict__ prod, const float* __restrict__ tproj,
                        const float* __restrict__ tbias, float* __restrict__ tw) {
  int b = blockIdx.x / T, u = blockIdx.x % T;
  int s = threadIdx.x;
  __shared__ float e[T];
  __shared__ float inv;
  if (s < T) {
    float acc = 0.f;
    for (int t = 0; t < T; ++t) {
      float p = prod[((size_t)b * T + t) * T + u] + tbias[t * T + u];
      acc += tproj[s * T + t] * (1.f / (1.f + __expf(-p)));
    }
    e[s] = acc;
  }
  __syncthreads();
  if (threadIdx.x == 0) {
    float m = -1e30f;
    for (int i = 0; i < T; ++i) m = fmaxf(m, e[i]);
    float sum = 0.f;
    for (int i = 0; i < T; ++i) { e[i] = __expf(e[i] - m); sum += e[i]; }
    inv = 1.f / sum;
  }
  __syncthreads();
  if (s < T) tw[((size_t)b * T + s) * T + u] = e[s] * inv;
}

// x_temp[b,n,f,u] = sum_t x[b,n,f,t]*tw[b,t,u]
__global__ __launch_bounds__(256) void k_xtemp(const float* __restrict__ x,
                                               const float* __restrict__ tw,
                                               float* __restrict__ xt) {
  int bn = blockIdx.x, b = bn >> 9, tid = threadIdx.x;
  __shared__ float xr[F * T];
  __shared__ float tws[T * T];
  const float* xp = x + (size_t)bn * F * T;
  for (int i = tid; i < F * T; i += 256) xr[i] = xp[i];
  const float* twp = tw + (size_t)b * T * T;
  for (int i = tid; i < T * T; i += 256) tws[i] = twp[i];
  __syncthreads();
  float* op = xt + (size_t)bn * F * T;
  for (int i = tid; i < F * T; i += 256) {
    int f = i / T, u = i % T;
    float s = 0.f;
    for (int t = 0; t < T; ++t) s += xr[f * T + t] * tws[t * T + u];
    op[i] = s;
  }
}

// =====================================================================
// Spatial attention front end
// =====================================================================
__global__ void k_sapre(const float* __restrict__ xt, const float* __restrict__ wa,
                        const float* __restrict__ wb, const float* __restrict__ wc,
                        float* __restrict__ lhs, float* __restrict__ rhs) {
  int bn = blockIdx.x, tid = threadIdx.x;   // blockDim = 32, tid == f
  const float* xp = xt + (size_t)bn * F * T;
  __shared__ float xa[F];
  float s = 0.f;
  for (int t = 0; t < T; ++t) s += xp[tid * T + t] * wa[t];
  xa[tid] = s;
  __syncthreads();
  if (tid < T) {
    float l = 0.f, r = 0.f;
    for (int f = 0; f < F; ++f) {
      l += xa[f] * wb[f * T + tid];
      r += wc[f] * xp[f * T + tid];
    }
    lhs[(size_t)bn * T + tid] = l;
    rhs[(size_t)bn * T + tid] = r;
  }
}

// attnbf[b,i,j] = bf16(sigmoid(lhs[b,i]·rhs[b,j] + sa_bias[i,j]))
__global__ __launch_bounds__(256) void k_attn(const float* __restrict__ lhs,
                                              const float* __restrict__ rhs,
                                              const float* __restrict__ bias,
                                              bf16* __restrict__ attnbf) {
  int b = blockIdx.z, n = blockIdx.y;
  int m = blockIdx.x * 256 + threadIdx.x;
  __shared__ float l[T];
  if (threadIdx.x < T) l[threadIdx.x] = lhs[((size_t)b * N + n) * T + threadIdx.x];
  __syncthreads();
  const float* rp = rhs + ((size_t)b * N + m) * T;
  float s = 0.f;
  for (int t = 0; t < T; ++t) s += l[t] * rp[t];
  s += bias[(size_t)n * N + m];
  attnbf[((size_t)b * N + n) * N + m] = (bf16)(1.f / (1.f + __expf(-s)));
}

// =====================================================================
// bf16 prep passes: convert / transpose operands into [out][K] row-major
// =====================================================================

// projbf = bf16(sa_proj)  (already [n][m])
__global__ void k_projbf(const float* __restrict__ in, bf16* __restrict__ out) {
  int i = blockIdx.x * blockDim.x + threadIdx.x;
  if (i < N * N) out[i] = (bf16)in[i];
}

// bf16 32x32 tiled transpose: out[b][j][i] = in[b][i][j], square NxN
__global__ __launch_bounds__(256) void k_tr_bf(const bf16* __restrict__ in,
                                               bf16* __restrict__ out) {
  __shared__ bf16 t[32][33];
  int b = blockIdx.z, i0 = blockIdx.y * 32, j0 = blockIdx.x * 32;
  int tid = threadIdx.x;
  const bf16* ip = in + (size_t)b * N * N;
  bf16* op = out + (size_t)b * N * N;
  for (int q = 0; q < 4; ++q) {
    int idx = tid + q * 256, r = idx >> 5, c = idx & 31;
    t[r][c] = ip[(size_t)(i0 + r) * N + j0 + c];
  }
  __syncthreads();
  for (int q = 0; q < 4; ++q) {
    int idx = tid + q * 256, r = idx >> 5, c = idx & 31;
    op[(size_t)(j0 + r) * N + i0 + c] = t[c][r];
  }
}

// XT[b][j][m] = bf16(x[b][m][j]),  j in [0,768)
__global__ __launch_bounds__(256) void k_trx(const float* __restrict__ in,
                                             bf16* __restrict__ out) {
  __shared__ bf16 t[32][33];
  int b = blockIdx.z, m0 = blockIdx.y * 32, j0 = blockIdx.x * 32;
  int tid = threadIdx.x;
  const float* ip = in + (size_t)b * N * (F * T);
  bf16* op = out + (size_t)b * N * (F * T);
  for (int q = 0; q < 4; ++q) {
    int idx = tid + q * 256, r = idx >> 5, c = idx & 31;
    t[r][c] = (bf16)ip[(size_t)(m0 + r) * (F * T) + j0 + c];
  }
  __syncthreads();
  for (int q = 0; q < 4; ++q) {
    int idx = tid + q * 256, r = idx >> 5, c = idx & 31;
    op[(size_t)(j0 + r) * N + m0 + c] = t[c][r];
  }
}

// ABk[b][n][m] = bf16(chebk[m][n] * spat[b][m][n])  (transposing pass)
__global__ __launch_bounds__(256) void k_abk(const float* __restrict__ chebk,
                                             const float* __restrict__ spat,
                                             bf16* __restrict__ out) {
  __shared__ bf16 t[32][33];
  int b = blockIdx.z, m0 = blockIdx.y * 32, n0 = blockIdx.x * 32;
  int tid = threadIdx.x;
  const float* sp = spat + (size_t)b * N * N;
  for (int q = 0; q < 4; ++q) {
    int idx = tid + q * 256, r = idx >> 5, c = idx & 31;
    size_t o = (size_t)(m0 + r) * N + n0 + c;
    t[r][c] = (bf16)(chebk[o] * sp[o]);
  }
  __syncthreads();
  bf16* op = out + (size_t)b * N * N;
  for (int q = 0; q < 4; ++q) {
    int idx = tid + q * 256, r = idx >> 5, c = idx & 31;
    op[(size_t)(n0 + r) * N + m0 + c] = t[c][r];
  }
}

// =====================================================================
// WMMA bf16 GEMM: C[b] = A(bf16 [512][512]) @ Bt(bf16 [ncols][512])^T
// 128x128 tile / 4-wave workgroup; each wave = 64x64 = 4x4 fragments of
// 16x16x32 (16 wmma per 16 ds_read_b128). Tiles staged with double-
// buffered global_load_async_to_lds_b128 (ASYNCcnt) — pure DMA staging.
// =====================================================================
__device__ inline bf16x16 load_frag(const bf16* row, int lg) {
  bf16x8 lo = *(const bf16x8*)(row + lg * 8);
  bf16x8 hi = *(const bf16x8*)(row + 16 + lg * 8);
  return __builtin_shufflevector(lo, hi, 0, 1, 2, 3, 4, 5, 6, 7,
                                 8, 9, 10, 11, 12, 13, 14, 15);
}

__global__ __launch_bounds__(128) void k_gemm(const bf16* __restrict__ A,
                                              const bf16* __restrict__ Bt,
                                              float* __restrict__ C,
                                              int ncols, int a_batched) {
  __shared__ __align__(16) bf16 As[2][128 * LDP];
  __shared__ __align__(16) bf16 Bs[2][128 * LDP];
  const int b   = blockIdx.z;
  const int n0  = blockIdx.y * 128;
  const int c0  = blockIdx.x * 128;
  const int tid = threadIdx.x;
  const int lane = tid & 31, wave = tid >> 5;
  const int wr = (wave >> 1) * 64, wc = (wave & 1) * 64;
  const int lg = lane >> 4, l15 = lane & 15;

  const bf16* Ab = A + (a_batched ? (size_t)b * N * KK : 0);
  const bf16* Bb = Bt + (size_t)b * (size_t)ncols * KK;
  float* Cb = C + (size_t)b * (size_t)N * ncols;

  f32x8 acc[4][4] = {};

  auto issue = [&](int buf, int kk) {
    const bf16* ga = Ab + (size_t)(n0 + tid) * KK + kk;
    const bf16* gb = Bb + (size_t)(c0 + tid) * KK + kk;
    unsigned la = (unsigned)(size_t)&As[buf][tid * LDP];
    unsigned lb = (unsigned)(size_t)&Bs[buf][tid * LDP];
    for (int c = 0; c < 4; ++c) {           // 32 bf16 = 4 x 16B per row
      async_copy16(la + c * 16, ga + c * 8);
      async_copy16(lb + c * 16, gb + c * 8);
    }
  };

  issue(0, 0);
  for (int s = 0; s < KK / 32; ++s) {
    const int cur = s & 1;
    wait_async0();            // current tiles landed (only this step in flight)
    __syncthreads();          // all waves' tiles visible; prev reads retired
    if (s + 1 < KK / 32) issue(1 - cur, (s + 1) * 32);  // overlap with compute
    bf16x16 af[4];
    for (int i = 0; i < 4; ++i)
      af[i] = load_frag(&As[cur][(wr + i * 16 + l15) * LDP], lg);
    for (int j = 0; j < 4; ++j) {
      bf16x16 bfr = load_frag(&Bs[cur][(wc + j * 16 + l15) * LDP], lg);
      for (int i = 0; i < 4; ++i)
        acc[i][j] = __builtin_amdgcn_wmma_f32_16x16x32_bf16(
            false, af[i], false, bfr, (short)0, acc[i][j], false, false);
    }
  }
  // D layout: VGPR e -> row = base + e + 8*(lane>=16); col = base + (lane&15)
  for (int i = 0; i < 4; ++i)
    for (int j = 0; j < 4; ++j) {
      int rbase = n0 + wr + i * 16 + lg * 8;
      int cbase = c0 + wc + j * 16 + l15;
      for (int e = 0; e < 8; ++e)
        Cb[(size_t)(rbase + e) * ncols + cbase] = acc[i][j][e];
    }
}

// column softmax over n for each (b,p)
__global__ __launch_bounds__(128) void k_spat(const float* __restrict__ S,
                                              float* __restrict__ spat) {
  int b = blockIdx.x >> 9, p = blockIdx.x & 511, tid = threadIdx.x;
  __shared__ float red[128];
  const float* sp = S + (size_t)b * N * N + p;
  float v[4], m = -1e30f;
  for (int i = 0; i < 4; ++i) {
    v[i] = sp[(size_t)(tid + 128 * i) * N];
    m = fmaxf(m, v[i]);
  }
  red[tid] = m; __syncthreads();
  for (int o = 64; o > 0; o >>= 1) {
    if (tid < o) red[tid] = fmaxf(red[tid], red[tid + o]);
    __syncthreads();
  }
  m = red[0]; __syncthreads();
  float sum = 0.f;
  for (int i = 0; i < 4; ++i) { v[i] = __expf(v[i] - m); sum += v[i]; }
  red[tid] = sum; __syncthreads();
  for (int o = 64; o > 0; o >>= 1) {
    if (tid < o) red[tid] += red[tid + o];
    __syncthreads();
  }
  float inv = 1.f / red[0];
  float* op = spat + (size_t)b * N * N + p;
  for (int i = 0; i < 4; ++i) op[(size_t)(tid + 128 * i) * N] = v[i] * inv;
}

// y[b,n,o,t] (+)= sum_f H[b,n,f,t]*Wk[f,o]
__global__ __launch_bounds__(256) void k_yacc(const float* __restrict__ H,
                                              const float* __restrict__ W,
                                              float* __restrict__ y, int first) {
  int bn = blockIdx.x, tid = threadIdx.x;
  __shared__ float hr[F * T];
  __shared__ float ws[F * GC];
  const float* hp = H + (size_t)bn * F * T;
  for (int i = tid; i < F * T; i += 256) hr[i] = hp[i];
  for (int i = tid; i < F * GC; i += 256) ws[i] = W[i];
  __syncthreads();
  float* yp = y + (size_t)bn * GC * T;
  for (int i = tid; i < GC * T; i += 256) {
    int o = i / T, t = i % T;
    float s = 0.f;
    for (int f = 0; f < F; ++f) s += hr[f * T + t] * ws[f * GC + o];
    if (first) yp[i] = s; else yp[i] += s;
  }
}

// relu(y) -> 1x3 time conv + 1x1 skip -> relu -> LayerNorm over TC -> out[b,n,c,t]
__global__ __launch_bounds__(256) void k_final(const float* __restrict__ y,
                                               const float* __restrict__ x,
                                               const float* __restrict__ tw,
                                               const float* __restrict__ tb,
                                               const float* __restrict__ sw,
                                               const float* __restrict__ sb,
                                               const float* __restrict__ lng,
                                               const float* __restrict__ lnb,
                                               float* __restrict__ out) {
  int bn = blockIdx.x, tid = threadIdx.x;
  __shared__ float gy[GC * T];
  __shared__ float xr[F * T];
  __shared__ float v[TC * T];
  __shared__ float mu[T], rs[T];
  const float* yp = y + (size_t)bn * GC * T;
  const float* xp = x + (size_t)bn * F * T;
  for (int i = tid; i < GC * T; i += 256) gy[i] = fmaxf(yp[i], 0.f);
  for (int i = tid; i < F * T; i += 256) xr[i] = xp[i];
  __syncthreads();
  for (int i = tid; i < TC * T; i += 256) {
    int c = i / T, t = i % T;
    float s = tb[c] + sb[c];
    const float* w = tw + c * GC * 3;
    for (int g = 0; g < GC; ++g) {
      float xm  = (t > 0)     ? gy[g * T + t - 1] : 0.f;
      float x0  = gy[g * T + t];
      float xp1 = (t < T - 1) ? gy[g * T + t + 1] : 0.f;
      s += w[g * 3] * xm + w[g * 3 + 1] * x0 + w[g * 3 + 2] * xp1;
    }
    const float* skw = sw + c * F;
    for (int f = 0; f < F; ++f) s += skw[f] * xr[f * T + t];
    v[i] = fmaxf(s, 0.f);
  }
  __syncthreads();
  if (tid < T) {
    float m = 0.f;
    for (int c = 0; c < TC; ++c) m += v[c * T + tid];
    m /= TC;
    float q = 0.f;
    for (int c = 0; c < TC; ++c) { float d = v[c * T + tid] - m; q += d * d; }
    mu[tid] = m;
    rs[tid] = rsqrtf(q / TC + 1e-5f);
  }
  __syncthreads();
  float* op = out + (size_t)bn * TC * T;
  for (int i = tid; i < TC * T; i += 256) {
    int c = i / T, t = i % T;
    op[i] = (v[i] - mu[t]) * rs[t] * lng[c] + lnb[c];
  }
}

// =====================================================================
extern "C" void kernel_launch(void* const* d_in, const int* in_sizes, int n_in,
                              void* d_out, int out_size, void* d_ws, size_t ws_size,
                              hipStream_t stream) {
  const float* x       = (const float*)d_in[0];
  const float* cheb    = (const float*)d_in[1];
  const float* w_cheb  = (const float*)d_in[2];
  const float* sa_wa   = (const float*)d_in[3];
  const float* sa_wb   = (const float*)d_in[4];
  const float* sa_wc   = (const float*)d_in[5];
  const float* sa_bias = (const float*)d_in[6];
  const float* sa_proj = (const float*)d_in[7];
  const float* ta_w1   = (const float*)d_in[8];
  const float* ta_w2   = (const float*)d_in[9];
  const float* ta_w3   = (const float*)d_in[10];
  const float* ta_bias = (const float*)d_in[11];
  const float* ta_proj = (const float*)d_in[12];
  const float* time_w  = (const float*)d_in[13];
  const float* time_b  = (const float*)d_in[14];
  const float* skip_w  = (const float*)d_in[15];
  const float* skip_b  = (const float*)d_in[16];
  const float* ln_g    = (const float*)d_in[17];
  const float* ln_b    = (const float*)d_in[18];
  float* out = (float*)d_out;

  char* base = (char*)d_ws;
  auto alloc = [&](size_t bytes) -> char* {
    char* r = base;
    base += (bytes + 255) & ~(size_t)255;
    return r;
  };
  float* xw1   = (float*)alloc((size_t)B * T * F * 4);
  float* lhs_t = (float*)alloc((size_t)B * T * N * 4);
  float* rhs_t = (float*)alloc((size_t)B * N * T * 4);
  float* prod  = (float*)alloc((size_t)B * T * T * 4);
  float* twb   = (float*)alloc((size_t)B * T * T * 4);
  float* xtmp  = (float*)alloc((size_t)B * N * F * T * 4);
  float* lhs_s = (float*)alloc((size_t)B * N * T * 4);
  float* rhs_s = (float*)alloc((size_t)B * N * T * 4);
  float* Sb    = (float*)alloc((size_t)B * N * N * 4);
  float* spat  = (float*)alloc((size_t)B * N * N * 4);
  float* Hb    = (float*)alloc((size_t)B * N * F * T * 4);
  float* yb    = (float*)alloc((size_t)B * N * GC * T * 4);
  bf16* projbf = (bf16*)alloc((size_t)N * N * 2);
  bf16* attnbf = (bf16*)alloc((size_t)B * N * N * 2);
  bf16* BTa    = (bf16*)alloc((size_t)B * N * N * 2);
  bf16* XT     = (bf16*)alloc((size_t)B * N * F * T * 2);
  bf16* ABk    = (bf16*)alloc((size_t)B * N * N * 2);

  // ---- temporal attention ----
  k_xw1  <<<(B * T * F + 255) / 256, 256, 0, stream>>>(x, ta_w1, xw1);
  k_rhs_t<<<(B * N * T + 255) / 256, 256, 0, stream>>>(x, ta_w3, rhs_t);
  k_lhs_t<<<(B * T * N + 255) / 256, 256, 0, stream>>>(xw1, ta_w2, lhs_t);
  k_prod <<<(B * T * T + 255) / 256, 256, 0, stream>>>(lhs_t, rhs_t, prod);
  k_tempw<<<B * T, 32, 0, stream>>>(prod, ta_proj, ta_bias, twb);
  k_xtemp<<<B * N, 256, 0, stream>>>(x, twb, xtmp);

  // independent bf16 prep (overlaps attention chain in stream order)
  k_projbf<<<(N * N + 255) / 256, 256, 0, stream>>>(sa_proj, projbf);
  k_trx   <<<dim3((F * T) / 32, N / 32, B), 256, 0, stream>>>(x, XT);

  // ---- spatial attention ----
  k_sapre<<<B * N, 32, 0, stream>>>(xtmp, sa_wa, sa_wb, sa_wc, lhs_s, rhs_s);
  k_attn <<<dim3(N / 256, N, B), 256, 0, stream>>>(lhs_s, rhs_s, sa_bias, attnbf);
  k_tr_bf<<<dim3(N / 32, N / 32, B), 256, 0, stream>>>(attnbf, BTa);
  k_gemm <<<dim3(N / 128, N / 128, B), 128, 0, stream>>>(projbf, BTa, Sb, N, 0);
  k_spat <<<B * N, 128, 0, stream>>>(Sb, spat);

  // ---- Chebyshev graph conv (k sequential; ABk/Hb reused) ----
  for (int k = 0; k < 3; ++k) {
    k_abk <<<dim3(N / 32, N / 32, B), 256, 0, stream>>>(cheb + (size_t)k * N * N,
                                                        spat, ABk);
    k_gemm<<<dim3((F * T) / 128, N / 128, B), 128, 0, stream>>>(ABk, XT, Hb,
                                                                F * T, 1);
    k_yacc<<<B * N, 256, 0, stream>>>(Hb, w_cheb + (size_t)k * F * GC, yb, k == 0);
  }

  // ---- time conv + skip + relu + layernorm ----
  k_final<<<B * N, 256, 0, stream>>>(yb, x, time_w, time_b, skip_w, skip_b,
                                     ln_g, ln_b, out);
}
